// MultiHeadAttention_8864812499477
// MI455X (gfx1250) — matrix operand order repaired and checked by reading, not compiled, not run on hardware
//
#include <hip/hip_runtime.h>

// ---------------------------------------------------------------------------
// MHA for MI455X (gfx1250): bf16 WMMA pipeline, flash-attention style,
// TDM (tensor_load_to_lds) staging for bf16 tiles, native bf16 converts.
// B=4 T=2048 C=512 H=8 D=64
// ---------------------------------------------------------------------------

typedef __attribute__((ext_vector_type(16))) __bf16 v16bf;
typedef __attribute__((ext_vector_type(8)))  __bf16 v8bf;
typedef __attribute__((ext_vector_type(8)))  float  v8f;
typedef __attribute__((ext_vector_type(4)))  unsigned u32x4;
typedef __attribute__((ext_vector_type(8)))  int      i32x8;
typedef __attribute__((ext_vector_type(4)))  int      i32x4;

#define DEV __device__ __forceinline__

DEV unsigned packbf(__bf16 a, __bf16 b) {
  union { __bf16 h[2]; unsigned u; } o;
  o.h[0] = a; o.h[1] = b;
  return o.u;
}
DEV unsigned pack2f(float a, float b) {        // f32 pair -> packed bf16
  return packbf((__bf16)a, (__bf16)b);
}

// LDS byte offset of a __shared__ object (flat LDS aperture: addr[31:0]=offset)
DEV unsigned lds_off(const void* p) { return (unsigned)(unsigned long long)p; }

// ---- Tensor Data Mover: 2D tile (bf16) global -> LDS, row-major -------------
// D# per CDNA5 ISA. group0: count=1, lds_addr,
// global_addr[56:0], type=2. group1: data_size=1(2B), tensor_dim0/1,
// tile_dim0/1, tensor_dim0_stride. Groups 2/3 zero (2D tensor).
// amdgpu-toolchain (clang-23) builtin arity: 6 args (g0,g1,g2,g3,g4,cpol).
DEV void tdm_load_2d_bf16(unsigned lds_addr, const void* gptr,
                          unsigned tensor_d0, unsigned tensor_d1,
                          unsigned tile_d0, unsigned tile_d1,
                          unsigned stride0) {
  unsigned long long ga = (unsigned long long)gptr;
  u32x4 g0;
  g0[0] = 1u;                                            // count=1 (valid D#)
  g0[1] = lds_addr;                                      // lds_addr [63:32]
  g0[2] = (unsigned)(ga & 0xffffffffu);                  // global_addr[31:0]
  g0[3] = (unsigned)((ga >> 32) & 0x01ffffffu) | (2u << 30);  // addr hi | type=2
  i32x8 g1;
  g1[0] = (int)(1u << 16);                               // data_size=1 (2 bytes)
  g1[1] = (int)(tensor_d0 << 16);                        // tensor_dim0[15:0]
  g1[2] = (int)((tensor_d0 >> 16) | (tensor_d1 << 16));  // dim0 hi | dim1 lo
  g1[3] = (int)((tensor_d1 >> 16) | (tile_d0 << 16));    // dim1 hi | tile_dim0
  g1[4] = (int)(tile_d1 & 0xffffu);                      // tile_dim1, tile_dim2=0
  g1[5] = (int)stride0;                                  // tensor_dim0_stride lo
  g1[6] = 0;
  g1[7] = 0;
  i32x4 z4; z4[0] = 0; z4[1] = 0; z4[2] = 0; z4[3] = 0;
  i32x8 z8;
#pragma unroll
  for (int i = 0; i < 8; ++i) z8[i] = 0;
  __builtin_amdgcn_tensor_load_to_lds(g0, g1, z4, z4, z8, 0);
}

// A-matrix fragment (16x32 bf16, MxK): lane L holds row M=L&15;
// VGPR0-3: K = khalf*8 + 0..7 ; VGPR4-7: K = 16 + khalf*8 + 0..7 (khalf=L>>4)
DEV v16bf load_fragA(const __bf16* row, int khalf) {
  v8bf lo = *(const v8bf*)(row + khalf * 8);
  v8bf hi = *(const v8bf*)(row + 16 + khalf * 8);
  v16bf r;
#pragma unroll
  for (int i = 0; i < 8; ++i) { r[i] = lo[i]; r[8 + i] = hi[i]; }
  return r;
}

// B-matrix fragment (32x16 bf16, KxN), LDS stored transposed: row = column N,
// contiguous over K. Lane L holds N=L&15; VGPR0-3: K = khalf*8 + 0..7 ; VGPR4-7: K = 16 + khalf*8 + 0..7.
DEV v16bf load_fragB(const __bf16* row, int khalf) {
  v8bf lo = *(const v8bf*)(row + khalf * 8);
  v8bf hi = *(const v8bf*)(row + 16 + khalf * 8);
  v16bf r;
#pragma unroll
  for (int i = 0; i < 8; ++i) { r[i] = lo[i]; r[8 + i] = hi[i]; }
  return r;
}

DEV v8f wmma_bf16(v16bf a, v16bf b, v8f c) {
  return __builtin_amdgcn_wmma_f32_16x16x32_bf16(false, a, false, b,
                                                 (short)0, c, false, false);
}

// ---------------------------------------------------------------------------
// Kernel 1: fused QKV projection.  q/k/v[b,h,t,d] = x[b,t,:] @ W{q,k,v}[h]
// grid = (M/128, H, 3), block = 256 (8 waves). Tile 128x64, BK=32.
// (f32 -> bf16 conversion happens while staging, so no TDM here.)
// ---------------------------------------------------------------------------
__global__ __launch_bounds__(256)
void qkv_proj_kernel(const float* __restrict__ x,
                     const float* __restrict__ Wq,
                     const float* __restrict__ Wk,
                     const float* __restrict__ Wv,
                     __bf16* __restrict__ qo,
                     __bf16* __restrict__ ko,
                     __bf16* __restrict__ vo) {
  constexpr int T = 2048, C = 512, D = 64, H = 8;
  constexpr int BM = 128, BK = 32;
  __shared__ __bf16 Als[BM][BK];   // x tile, row-major (row stride 64B)
  __shared__ __bf16 Bls[D][BK];    // W tile transposed: Bls[d][k]

  const int tid = threadIdx.x, wave = tid >> 5, lane = tid & 31;
  const int ll = lane & 15, lh = lane >> 4;
  const int m0 = blockIdx.x * BM;
  const int h  = blockIdx.y;
  const int wc = blockIdx.z;
  const float* W = (wc == 0) ? Wq : (wc == 1) ? Wk : Wv;
  __bf16* outp   = (wc == 0) ? qo : (wc == 1) ? ko : vo;
  W += (size_t)h * C * D;

  v8f acc[4];
#pragma unroll
  for (int i = 0; i < 4; ++i)
#pragma unroll
    for (int j = 0; j < 8; ++j) acc[i][j] = 0.f;

  for (int kb = 0; kb < C; kb += BK) {
    // stage A: 128x32 f32 -> bf16 packed u32 stores
    {
      int row = tid >> 1, c0 = (tid & 1) * 16;
      const float* src = x + (size_t)(m0 + row) * C + kb + c0;
      unsigned* dst = (unsigned*)&Als[row][c0];
#pragma unroll
      for (int i = 0; i < 8; ++i) dst[i] = pack2f(src[2 * i], src[2 * i + 1]);
    }
    // stage B transposed: thread owns one d, 8 consecutive k (packed stores)
    {
      int d = tid >> 2, k0 = (tid & 3) * 8;
      const float* src = W + (size_t)(kb + k0) * D + d;   // stride D over k
      unsigned* dst = (unsigned*)&Bls[d][k0];
#pragma unroll
      for (int i = 0; i < 4; ++i)
        dst[i] = pack2f(src[(2 * i) * D], src[(2 * i + 1) * D]);
    }
    __syncthreads();
    v16bf a = load_fragA(&Als[wave * 16 + ll][0], lh);
#pragma unroll
    for (int nt = 0; nt < 4; ++nt) {
      v16bf b = load_fragB(&Bls[nt * 16 + ll][0], lh);
      acc[nt] = wmma_bf16(a, b, acc[nt]);
    }
    __syncthreads();
  }

  // store: element (m=r+8*lh, n=nt*16+ll); layout [B,H,T,D]
#pragma unroll
  for (int nt = 0; nt < 4; ++nt)
#pragma unroll
    for (int r = 0; r < 8; ++r) {
      int g = m0 + wave * 16 + r + 8 * lh;     // global x-row
      int b = g >> 11, t = g & 2047;
      outp[(((size_t)(b * H + h)) * T + t) * D + nt * 16 + ll] =
          (__bf16)acc[nt][r];
    }
}

// ---------------------------------------------------------------------------
// Kernel 2: flash attention per (b,h). 128 queries/block, 64-key tiles.
// Q and K tiles staged by the Tensor Data Mover; V staged manually (needs
// transpose). grid = (T/128, B*H), block = 256.
// ---------------------------------------------------------------------------
__global__ __launch_bounds__(256)
void attn_kernel(const __bf16* __restrict__ q,
                 const __bf16* __restrict__ k,
                 const __bf16* __restrict__ v,
                 __bf16* __restrict__ o) {
  constexpr int T = 2048, D = 64, H = 8;
  constexpr int BQ = 128, BKEY = 64;
  __shared__ __bf16 Qls[BQ][D];        // 16 KB (TDM target)
  __shared__ __bf16 Kls[BKEY][D];      // 8 KB  (TDM target; row=key, contig d)
  __shared__ __bf16 Vls[D][BKEY];      // 8 KB  (transposed: row=d, contig key)
  __shared__ __bf16 Pls[8][16][BKEY];  // 16 KB (per-wave P staging)

  const int tid = threadIdx.x, wave = tid >> 5, lane = tid & 31;
  const int ll = lane & 15, lh = lane >> 4;
  const int q0 = blockIdx.x * BQ;
  const int bh = blockIdx.y;
  const size_t base = (size_t)bh * T * D;
  const float scale = 0.044194173824159216f;  // C^-0.5, C=512

  // stage Q once via TDM: 128x64 tile of the (b,h) 2048x64 tensor
  if (wave == 0) {
    tdm_load_2d_bf16(lds_off(&Qls[0][0]), q + base + (size_t)q0 * D,
                     /*tensor_d0=*/D, /*tensor_d1=*/T,
                     /*tile_d0=*/D, /*tile_d1=*/BQ, /*stride0=*/D);
    __builtin_amdgcn_s_wait_tensorcnt(0);
  }
  __syncthreads();

  const __bf16* qrow = &Qls[wave * 16 + ll][0];
  v16bf qa0 = load_fragA(qrow, lh);        // d 0..31
  v16bf qa1 = load_fragA(qrow + 32, lh);   // d 32..63

  float mi[8], li[8];
  v8f oacc[4];
#pragma unroll
  for (int r = 0; r < 8; ++r) { mi[r] = -3.402823466e38f; li[r] = 0.f; }
#pragma unroll
  for (int i = 0; i < 4; ++i)
#pragma unroll
    for (int j = 0; j < 8; ++j) oacc[i][j] = 0.f;

  for (int j0 = 0; j0 < T; j0 += BKEY) {
    // K tile via TDM: 64x64
    if (wave == 0)
      tdm_load_2d_bf16(lds_off(&Kls[0][0]), k + base + (size_t)j0 * D,
                       D, T, D, BKEY, D);
    // V tile transposed, packed u32 LDS stores (thread owns one d)
    {
      int d = tid >> 2, key0 = (tid & 3) * 16;
      const __bf16* src = v + base + (size_t)(j0 + key0) * D + d;
      unsigned* dst = (unsigned*)&Vls[d][key0];
#pragma unroll
      for (int i = 0; i < 8; ++i)
        dst[i] = packbf(src[(2 * i) * D], src[(2 * i + 1) * D]);
      if (j0 + BKEY < T)  // prefetch next V tile (global_prefetch_b8)
        __builtin_prefetch(src + (size_t)BKEY * D, 0, 0);
    }
    if (wave == 0) __builtin_amdgcn_s_wait_tensorcnt(0);
    __syncthreads();

    // S = Q K^T  (16 x 64 per wave)
    v8f s[4];
#pragma unroll
    for (int nt = 0; nt < 4; ++nt) {
#pragma unroll
      for (int e = 0; e < 8; ++e) s[nt][e] = 0.f;
      const __bf16* krow = &Kls[nt * 16 + ll][0];
      s[nt] = wmma_bf16(qa0, load_fragB(krow, lh), s[nt]);
      s[nt] = wmma_bf16(qa1, load_fragB(krow + 32, lh), s[nt]);
    }

    // online softmax (row m = r + 8*lh; rows live in 16-lane halves)
    float mnew[8], alpha[8];
#pragma unroll
    for (int r = 0; r < 8; ++r) {
      float mx = -3.402823466e38f;
#pragma unroll
      for (int nt = 0; nt < 4; ++nt) mx = fmaxf(mx, s[nt][r]);
      mx *= scale;
#pragma unroll
      for (int msk = 1; msk < 16; msk <<= 1)
        mx = fmaxf(mx, __shfl_xor(mx, msk, 32));
      mnew[r]  = fmaxf(mi[r], mx);
      alpha[r] = __expf(mi[r] - mnew[r]);
    }
#pragma unroll
    for (int r = 0; r < 8; ++r) {
      int m = r + 8 * lh;
      float p0 = __expf(s[0][r] * scale - mnew[r]);
      float p1 = __expf(s[1][r] * scale - mnew[r]);
      float p2 = __expf(s[2][r] * scale - mnew[r]);
      float p3 = __expf(s[3][r] * scale - mnew[r]);
      float sum = (p0 + p1) + (p2 + p3);
      Pls[wave][m][0 * 16 + ll] = (__bf16)p0;
      Pls[wave][m][1 * 16 + ll] = (__bf16)p1;
      Pls[wave][m][2 * 16 + ll] = (__bf16)p2;
      Pls[wave][m][3 * 16 + ll] = (__bf16)p3;
#pragma unroll
      for (int msk = 1; msk < 16; msk <<= 1) sum += __shfl_xor(sum, msk, 32);
      li[r] = li[r] * alpha[r] + sum;
      mi[r] = mnew[r];
    }
#pragma unroll
    for (int dt = 0; dt < 4; ++dt)
#pragma unroll
      for (int r = 0; r < 8; ++r) oacc[dt][r] *= alpha[r];

    // make this wave's P stores visible to its own fragment loads
    asm volatile("s_wait_dscnt 0" ::: "memory");

    // O += P V
    const __bf16* prow = &Pls[wave][ll][0];
    v16bf pa0 = load_fragA(prow, lh);        // keys 0..31
    v16bf pa1 = load_fragA(prow + 32, lh);   // keys 32..63
#pragma unroll
    for (int dt = 0; dt < 4; ++dt) {
      const __bf16* vrow = &Vls[dt * 16 + ll][0];
      oacc[dt] = wmma_bf16(pa0, load_fragB(vrow, lh), oacc[dt]);
      oacc[dt] = wmma_bf16(pa1, load_fragB(vrow + 32, lh), oacc[dt]);
    }
    __syncthreads();
  }

  // normalize + store as bf16, concat-head layout [B, T, H*D]
  const int b = bh >> 3, h = bh & 7;
#pragma unroll
  for (int dt = 0; dt < 4; ++dt)
#pragma unroll
    for (int r = 0; r < 8; ++r) {
      int t = q0 + wave * 16 + r + 8 * lh;
      float val = oacc[dt][r] / li[r];
      o[((size_t)(b * T + t)) * (H * D) + h * D + dt * 16 + ll] = (__bf16)val;
    }
}

// ---------------------------------------------------------------------------
// Kernel 3: output projection [8192,512] @ Wo[512,512] + bo, fp32 out.
// A tile (already bf16) staged by TDM; Wo converted while staging.
// grid = (M/128, N/64), block = 256.
// ---------------------------------------------------------------------------
__global__ __launch_bounds__(256)
void out_proj_kernel(const __bf16* __restrict__ oin,
                     const float* __restrict__ Wo,
                     const float* __restrict__ bo,
                     float* __restrict__ out) {
  constexpr int M = 8192, N = 512, K = 512;
  constexpr int BM = 128, BK = 32;
  __shared__ __bf16 Als[BM][BK];
  __shared__ __bf16 Bls[64][BK];
  const int tid = threadIdx.x, wave = tid >> 5, lane = tid & 31;
  const int ll = lane & 15, lh = lane >> 4;
  const int m0 = blockIdx.x * BM, nb = blockIdx.y * 64;

  v8f acc[4];
#pragma unroll
  for (int i = 0; i < 4; ++i)
#pragma unroll
    for (int j = 0; j < 8; ++j) acc[i][j] = 0.f;

  for (int kb = 0; kb < K; kb += BK) {
    // A tile 128x32 via TDM from bf16 [M,K] tensor
    if (wave == 0)
      tdm_load_2d_bf16(lds_off(&Als[0][0]), oin + (size_t)m0 * K + kb,
                       /*tensor_d0=*/K, /*tensor_d1=*/M,
                       /*tile_d0=*/BK, /*tile_d1=*/BM, /*stride0=*/K);
    // B tile transposed: thread owns one n, 8 consecutive k (packed stores)
    {
      int d = tid >> 2, k0 = (tid & 3) * 8;
      const float* src = Wo + (size_t)(kb + k0) * N + nb + d;
      unsigned* dst = (unsigned*)&Bls[d][k0];
#pragma unroll
      for (int i = 0; i < 4; ++i)
        dst[i] = pack2f(src[(2 * i) * N], src[(2 * i + 1) * N]);
    }
    if (wave == 0) __builtin_amdgcn_s_wait_tensorcnt(0);
    __syncthreads();
    v16bf a = load_fragA(&Als[wave * 16 + ll][0], lh);
#pragma unroll
    for (int nt = 0; nt < 4; ++nt)
      acc[nt] = wmma_bf16(a, load_fragB(&Bls[nt * 16 + ll][0], lh), acc[nt]);
    __syncthreads();
  }

#pragma unroll
  for (int nt = 0; nt < 4; ++nt) {
    float bias = bo[nb + nt * 16 + ll];
#pragma unroll
    for (int r = 0; r < 8; ++r) {
      int m = m0 + wave * 16 + r + 8 * lh;
      out[(size_t)m * N + nb + nt * 16 + ll] = acc[nt][r] + bias;
    }
  }
}

// ---------------------------------------------------------------------------
extern "C" void kernel_launch(void* const* d_in, const int* in_sizes, int n_in,
                              void* d_out, int out_size, void* d_ws, size_t ws_size,
                              hipStream_t stream) {
  (void)in_sizes; (void)n_in; (void)out_size; (void)ws_size;
  const float* x  = (const float*)d_in[0];
  const float* Wq = (const float*)d_in[1];
  const float* Wk = (const float*)d_in[2];
  const float* Wv = (const float*)d_in[3];
  const float* Wo = (const float*)d_in[4];
  const float* bo = (const float*)d_in[5];
  float* out = (float*)d_out;

  constexpr size_t QKV_ELEMS = (size_t)4 * 8 * 2048 * 64;  // B*H*T*D
  __bf16* qw = (__bf16*)d_ws;
  __bf16* kw = qw + QKV_ELEMS;
  __bf16* vw = kw + QKV_ELEMS;
  __bf16* ow = vw + QKV_ELEMS;  // [B, T, H*D] bf16

  qkv_proj_kernel<<<dim3(64, 8, 3), 256, 0, stream>>>(x, Wq, Wk, Wv, qw, kw, vw);
  attn_kernel<<<dim3(16, 32), 256, 0, stream>>>(qw, kw, vw, ow);
  out_proj_kernel<<<dim3(64, 8), 256, 0, stream>>>(ow, Wo, bo, out);
}
